// SupervisedGraphSage_69157563400577
// MI455X (gfx1250) — matrix operand order repaired and checked by reading, not compiled, not run on hardware
//
#include <hip/hip_runtime.h>

// ---- vector types ----
typedef __attribute__((ext_vector_type(16))) _Float16 v16h;
typedef __attribute__((ext_vector_type(8)))  _Float16 v8h;
typedef __attribute__((ext_vector_type(8)))  float    v8f;
typedef __attribute__((ext_vector_type(4)))  float    v4f;

#define DFEAT 128      // D_FEAT
#define DIN   256      // 2*D_FEAT (W1 K-dim)
#define DOUTN 256      // D_OUT (W3 K-dim)
#define NCLS  128      // N_CLASSES
#define DEG   16
#define MROWS 32       // rows (batch elements) per workgroup

#define EMB_STRIDE 264 // 256 + 8 f16 pad (row = 528 B, 16B aligned, breaks bank alignment)
#define H_STRIDE   264
#define L_STRIDE   136 // 128 + 8 f32 pad

// A-fragment (16-bit A 16x32, ISA 7.12.2): lanes 0-15 hold K 0-7 & 16-23,
// lanes 16-31 hold K 8-15 & 24-31. p points at row start + 32*ks halves.
static __device__ __forceinline__ v16h frag_a(const _Float16* p, int kh) {
  const v8h* p8 = (const v8h*)p;
  v8h lo = p8[kh];        // K: 8*kh .. 8*kh+7
  v8h hi = p8[2 + kh];    // K: 16+8*kh .. 23+8*kh
  return __builtin_shufflevector(lo, hi, 0,1,2,3,4,5,6,7,8,9,10,11,12,13,14,15);
}
// B-fragment (dense 16-bit B 32x16): lanes 0-15 hold K 0-7 & 16-23, lanes 16-31 K 8-15 & 24-31,
// column n = lane&15. p points at W row n (K-contiguous) + 32*ks halves.
static __device__ __forceinline__ v16h frag_b(const _Float16* p, int kh) {
  const v8h* p8 = (const v8h*)p;
  v8h lo = p8[kh];
  v8h hi = p8[2 + kh];
  return __builtin_shufflevector(lo, hi, 0,1,2,3,4,5,6,7,8,9,10,11,12,13,14,15);
}
static __device__ __forceinline__ v8f wmma16(v16h a, v16h b, v8f c) {
  return __builtin_amdgcn_wmma_f32_16x16x32_f16(false, a, false, b, (short)0, c,
                                                false, false);
}

// ---- weight fp32 -> f16 conversion (runs every launch; deterministic) ----
__global__ void sage_cvt_weights(const float* __restrict__ W1,
                                 const float* __restrict__ W3,
                                 _Float16* __restrict__ W1h,
                                 _Float16* __restrict__ W3h) {
  int i = blockIdx.x * blockDim.x + threadIdx.x;
  if (i < DIN * DOUTN) {                          // 65536
    W1h[i] = (_Float16)W1[i];
  } else if (i < DIN * DOUTN + NCLS * DOUTN) {    // +32768
    int j = i - DIN * DOUTN;
    W3h[j] = (_Float16)W3[j];
  }
}

// ---- fused gather-mean + MLP + L2-normalize, 32 batch rows per block ----
extern "C" __global__ void __launch_bounds__(128)
sage_fused(const int* __restrict__ gInp, const int* __restrict__ gNbr,
           const float* __restrict__ gFeat,
           const _Float16* __restrict__ W1h, const float* __restrict__ gB1,
           const _Float16* __restrict__ W3h, const float* __restrict__ gB3,
           float* __restrict__ gOut, int Bn)
{
  __shared__ __align__(16) _Float16 emb [MROWS * EMB_STRIDE];
  __shared__ __align__(16) _Float16 hbuf[MROWS * H_STRIDE];
  __shared__ __align__(16) float    lgt [MROWS * L_STRIDE];
  __shared__ float invn[MROWS];

  const int tid  = threadIdx.x;
  const int lane = tid & 31;
  const int w    = tid >> 5;      // wave 0..3
  const int ln   = lane & 15;
  const int kh   = lane >> 4;     // K-half (A/B) & M-half (C/D) selector
  const int m0   = blockIdx.x * MROWS;

  // ---------- Stage 1: gather self + neighbor mean -> f16 emb in LDS ----------
  {
    const int r  = tid >> 2;           // row 0..31
    const int c0 = (tid & 3) << 5;     // 32-feature chunk
    int gb = m0 + r; if (gb >= Bn) gb = Bn - 1;

    const v4f* s4 = (const v4f*)(gFeat + (long)gInp[gb] * DFEAT + c0);
    _Float16* es = emb + r * EMB_STRIDE + c0;
    #pragma unroll
    for (int q = 0; q < 8; ++q) {
      v4f v = s4[q];
      #pragma unroll
      for (int i = 0; i < 4; ++i) es[4 * q + i] = (_Float16)v[i];
    }

    v4f acc[8] = {};
    const long eb = (long)gb * DEG;
    #pragma unroll 4
    for (int d = 0; d < DEG; ++d) {
      const v4f* n4 = (const v4f*)(gFeat + (long)gNbr[eb + d] * DFEAT + c0);
      #pragma unroll
      for (int q = 0; q < 8; ++q) acc[q] += n4[q];
    }
    _Float16* en = emb + r * EMB_STRIDE + DFEAT + c0;
    #pragma unroll
    for (int q = 0; q < 8; ++q)
      #pragma unroll
      for (int i = 0; i < 4; ++i)
        en[4 * q + i] = (_Float16)(acc[q][i] * 0.0625f);
  }
  __syncthreads();

  // ---------- Stage 2: h = relu(emb @ W1^T + b1) -> f16 hbuf in LDS ----------
  // Each wave: 4 N-tiles (64 cols) x 2 M-tiles; B-fragments reused across M.
  {
    const _Float16* ar0 = emb + ln * EMB_STRIDE;           // M-tile 0
    const _Float16* ar1 = emb + (ln + 16) * EMB_STRIDE;    // M-tile 1
    const int nb = (w << 6) + ln;
    const _Float16* wr0 = W1h + (long)(nb +  0) * DIN;
    const _Float16* wr1 = W1h + (long)(nb + 16) * DIN;
    const _Float16* wr2 = W1h + (long)(nb + 32) * DIN;
    const _Float16* wr3 = W1h + (long)(nb + 48) * DIN;
    v8f c00 = {}, c01 = {}, c02 = {}, c03 = {};
    v8f c10 = {}, c11 = {}, c12 = {}, c13 = {};
    #pragma unroll
    for (int ks = 0; ks < 8; ++ks) {                       // K = 256, step 32
      v16h a0 = frag_a(ar0 + ks * 32, kh);
      v16h a1 = frag_a(ar1 + ks * 32, kh);
      v16h b0 = frag_b(wr0 + ks * 32, kh);
      c00 = wmma16(a0, b0, c00);  c10 = wmma16(a1, b0, c10);
      v16h b1 = frag_b(wr1 + ks * 32, kh);
      c01 = wmma16(a0, b1, c01);  c11 = wmma16(a1, b1, c11);
      v16h b2 = frag_b(wr2 + ks * 32, kh);
      c02 = wmma16(a0, b2, c02);  c12 = wmma16(a1, b2, c12);
      v16h b3 = frag_b(wr3 + ks * 32, kh);
      c03 = wmma16(a0, b3, c03);  c13 = wmma16(a1, b3, c13);
    }
    auto store_h = [&](v8f c, int n, int mt) {
      const float bias = gB1[n];
      #pragma unroll
      for (int v = 0; v < 8; ++v) {
        float hv = fmaxf(c[v] + bias, 0.0f);               // ReLU
        hbuf[(v + (kh << 3) + (mt << 4)) * H_STRIDE + n] = (_Float16)hv;
      }
    };
    store_h(c00, nb +  0, 0);  store_h(c01, nb + 16, 0);
    store_h(c02, nb + 32, 0);  store_h(c03, nb + 48, 0);
    store_h(c10, nb +  0, 1);  store_h(c11, nb + 16, 1);
    store_h(c12, nb + 32, 1);  store_h(c13, nb + 48, 1);
  }
  __syncthreads();

  // ---------- Stage 3: logits = h @ W3^T + b3 -> f32 lgt in LDS ----------
  // Each wave: 2 N-tiles (32 cols) x 2 M-tiles.
  {
    const _Float16* ar0 = hbuf + ln * H_STRIDE;
    const _Float16* ar1 = hbuf + (ln + 16) * H_STRIDE;
    const int nb = (w << 5) + ln;
    const _Float16* p0 = W3h + (long)(nb +  0) * DOUTN;
    const _Float16* p1 = W3h + (long)(nb + 16) * DOUTN;
    v8f d00 = {}, d01 = {}, d10 = {}, d11 = {};
    #pragma unroll
    for (int ks = 0; ks < 8; ++ks) {
      v16h a0 = frag_a(ar0 + ks * 32, kh);
      v16h a1 = frag_a(ar1 + ks * 32, kh);
      v16h b0 = frag_b(p0 + ks * 32, kh);
      d00 = wmma16(a0, b0, d00);  d10 = wmma16(a1, b0, d10);
      v16h b1 = frag_b(p1 + ks * 32, kh);
      d01 = wmma16(a0, b1, d01);  d11 = wmma16(a1, b1, d11);
    }
    auto store_l = [&](v8f c, int n, int mt) {
      const float bias = gB3[n];
      #pragma unroll
      for (int v = 0; v < 8; ++v)
        lgt[(v + (kh << 3) + (mt << 4)) * L_STRIDE + n] = c[v] + bias;
    };
    store_l(d00, nb +  0, 0);  store_l(d01, nb + 16, 0);
    store_l(d10, nb +  0, 1);  store_l(d11, nb + 16, 1);
  }
  __syncthreads();

  // ---------- Stage 4: row L2 norms and normalized store ----------
  if (tid < MROWS) {
    float s = 0.0f;
    #pragma unroll 8
    for (int j = 0; j < NCLS; ++j) {
      float x = lgt[tid * L_STRIDE + j];
      s += x * x;
    }
    invn[tid] = 1.0f / fmaxf(sqrtf(s), 1e-12f);
  }
  __syncthreads();
  {
    const int r  = tid >> 2;
    const int c0 = (tid & 3) << 5;
    const int gb = m0 + r;
    if (gb < Bn) {
      const float s = invn[r];
      float* o = gOut + (long)gb * NCLS + c0;
      #pragma unroll
      for (int i = 0; i < 32; ++i) o[i] = lgt[r * L_STRIDE + c0 + i] * s;
    }
  }
}

extern "C" void kernel_launch(void* const* d_in, const int* in_sizes, int n_in,
                              void* d_out, int out_size, void* d_ws, size_t ws_size,
                              hipStream_t stream) {
  (void)n_in; (void)out_size; (void)ws_size;
  const int*   inputs = (const int*)  d_in[0];
  const int*   nbr    = (const int*)  d_in[1];
  // d_in[2] segment_ids: implicit (16 contiguous entries per segment)
  const float* feat   = (const float*)d_in[3];
  const float* W1     = (const float*)d_in[4];
  const float* b1     = (const float*)d_in[5];
  const float* W3     = (const float*)d_in[6];
  const float* b3     = (const float*)d_in[7];
  float* out = (float*)d_out;
  const int B = in_sizes[0];

  _Float16* W1h = (_Float16*)d_ws;               // 256*256 f16 = 128 KB
  _Float16* W3h = W1h + (long)DIN * DOUTN;       // 128*256 f16 =  64 KB

  const int cvtN = DIN * DOUTN + NCLS * DOUTN;   // 98304
  sage_cvt_weights<<<(cvtN + 255) / 256, 256, 0, stream>>>(W1, W3, W1h, W3h);

  const int nblk = (B + MROWS - 1) / MROWS;      // 1250 for B=40000
  sage_fused<<<nblk, 128, 0, stream>>>(inputs, nbr, feat, W1h, b1, W3h, b3, out, B);
}
